// FilterStage3x3_52888227282951
// MI455X (gfx1250) — matrix operand order repaired and checked by reading, not compiled, not run on hardware
//
#include <hip/hip_runtime.h>

typedef __attribute__((ext_vector_type(16))) _Float16 v16h;
typedef __attribute__((ext_vector_type(8)))  float    v8f;

namespace {
constexpr int kB = 2, kF = 128, kC = 3, kH = 720, kW = 1280;
constexpr int kHW = kH * kW;            // 921600
constexpr int kNPix = kB * kHW;         // 1843200
constexpr int kPixPerWave  = 64;        // 2 pairs of (2 x 16-pixel WMMA tiles)
constexpr int kPixPerBlock = 8 * kPixPerWave;  // 512
constexpr unsigned kPlaneB = (unsigned)kHW * 4u;  // byte stride between feature planes
}

__device__ __forceinline__ float ldf(const float* p, unsigned byteOff) {
  return *(const float*)((const char*)p + (size_t)byteOff);
}

// One 16-pixel WMMA column tile: accumulate logits D = Wc(16x128) * f(128x16).
__device__ __forceinline__ v8f accum_tile(const float* __restrict__ f,
                                          unsigned base, int hi,
                                          const v16h A[4]) {
  v8f acc = {};
  #pragma unroll
  for (int c4 = 0; c4 < 4; ++c4) {
    v16h Bm;
    #pragma unroll
    for (int v = 0; v < 8; ++v) {
      const unsigned k0 = (unsigned)(32 * c4 + ((v >= 4) ? 16 : 0) + hi * 8 + 2 * (v & 3));
      const float a0 = ldf(f, base + k0 * kPlaneB);
      const float a1 = ldf(f, base + (k0 + 1u) * kPlaneB);
      Bm[2 * v]     = (_Float16)a0;
      Bm[2 * v + 1] = (_Float16)a1;
    }
    acc = __builtin_amdgcn_wmma_f32_16x16x32_f16(false, A[c4], false, Bm,
                                                 (short)0, acc, false, false);
  }
  return acc;
}

// Fused: 1x1 conv (WMMA f16->f32) + softmax(9) + per-pixel 3x3 filter of x.
__global__ __launch_bounds__(256) void filter_stage3x3_wmma(
    const float* __restrict__ f, const float* __restrict__ x,
    const float* __restrict__ Wc, const float* __restrict__ bc,
    float* __restrict__ out)
{
  const int lane = threadIdx.x & 31;
  const int wave = threadIdx.x >> 5;
  const int hi   = lane >> 4;   // wave half
  const int lo   = lane & 15;

  // ---- A matrix: Wc as 16x128 f16 (rows 9..15 zero), four 16x32 chunks ----
  // ISA 7.12.2 16-bit A layout: lanes 0-15 row M=lo hold K {0..7,16..23},
  // lanes 16-31 same row hold K {8..15,24..31} per 32-wide chunk.
  v16h A[4];
  #pragma unroll
  for (int c4 = 0; c4 < 4; ++c4) {
    #pragma unroll
    for (int v = 0; v < 8; ++v) {
      #pragma unroll
      for (int s = 0; s < 2; ++s) {
        const int k = 32 * c4 + ((v >= 4) ? 16 : 0) + hi * 8 + 2 * (v & 3) + s;
        const float val = (lo < 9) ? Wc[lo * kF + k] : 0.0f;
        A[c4][2 * v + s] = (_Float16)val;
      }
    }
  }

  const float b0 = bc[0], b1 = bc[1], b2 = bc[2], b3 = bc[3], b4 = bc[4],
              b5 = bc[5], b6 = bc[6], b7 = bc[7], b8 = bc[8];
  const float bcv[9] = {b0, b1, b2, b3, b4, b5, b6, b7, b8};

  const int waveBase = blockIdx.x * kPixPerBlock + wave * kPixPerWave;

  #pragma unroll
  for (int t = 0; t < 2; ++t) {
    const int p0 = waveBase + t * 32;        // 32 consecutive pixels, one row
    const int b  = p0 / kHW;
    const int r  = p0 - b * kHW;
    const int h  = r / kW;
    const int w0 = r - h * kW;               // multiple of 32

    // Byte offset of pixel (b, h, w0+lo) in feature plane 0 of f.
    const unsigned rowByte =
        ((unsigned)(b * kF) * (unsigned)kHW + (unsigned)(h * kW + w0)) * 4u;
    const unsigned base0 = rowByte + (unsigned)lo * 4u;  // tile0: pixels p0+0..15
    const unsigned base1 = base0 + 64u;                  // tile1: pixels p0+16..31

    const v8f acc0 = accum_tile(f, base0, hi, A);
    const v8f acc1 = accum_tile(f, base1, hi, A);

    // Redistribute logits so lane L owns pixel P = p0 + L.
    // D layout: lanes 0-15 hold taps 0-7 of col=lane; lanes 16-31 hold taps 8-15.
    float l[9];
    const float t8lo = __shfl_xor(acc0[0], 16, 32);  // tap8 for lanes < 16
    #pragma unroll
    for (int i = 0; i < 8; ++i) {
      const float thi = __shfl_xor(acc1[i], 16, 32); // taps 0-7 for lanes >= 16
      l[i] = ((lane < 16) ? acc0[i] : thi) + bcv[i];
    }
    l[8] = ((lane < 16) ? t8lo : acc1[0]) + bcv[8];

    // Softmax over 9 taps.
    float mx = l[0];
    #pragma unroll
    for (int i = 1; i < 9; ++i) mx = fmaxf(mx, l[i]);
    float e[9];
    float sum = 0.0f;
    #pragma unroll
    for (int i = 0; i < 9; ++i) { e[i] = __expf(l[i] - mx); sum += e[i]; }
    const float inv = 1.0f / sum;

    // Per-pixel 3x3 filter of x (zero-padded borders via zeroed weights).
    const int w = w0 + lane;
    #pragma unroll
    for (int c = 0; c < kC; ++c) {
      const unsigned chanOff = (unsigned)((b * kC + c) * kH);
      float a = 0.0f;
      #pragma unroll
      for (int dy = 0; dy < 3; ++dy) {
        const int hh = h + dy - 1;
        const int hc = (hh < 0) ? 0 : ((hh >= kH) ? kH - 1 : hh);
        const bool okh = ((unsigned)hh < (unsigned)kH);
        #pragma unroll
        for (int dx = 0; dx < 3; ++dx) {
          const int ww = w + dx - 1;
          const int wc = (ww < 0) ? 0 : ((ww >= kW) ? kW - 1 : ww);
          const bool ok = okh && ((unsigned)ww < (unsigned)kW);
          const float xv = x[(size_t)(chanOff + (unsigned)hc) * kW + wc];
          const float wt = ok ? (e[3 * dy + dx] * inv) : 0.0f;
          a = fmaf(wt, xv, a);
        }
      }
      out[(size_t)(chanOff + (unsigned)h) * kW + w] = a;
    }
  }
}

extern "C" void kernel_launch(void* const* d_in, const int* in_sizes, int n_in,
                              void* d_out, int out_size, void* d_ws, size_t ws_size,
                              hipStream_t stream) {
  (void)in_sizes; (void)n_in; (void)out_size; (void)d_ws; (void)ws_size;
  const float* f  = (const float*)d_in[0];
  const float* x  = (const float*)d_in[1];
  const float* Wc = (const float*)d_in[2];
  const float* bc = (const float*)d_in[3];
  float* out = (float*)d_out;
  const int grid = kNPix / kPixPerBlock;  // 3600 blocks x 256 threads (8 waves)
  filter_stage3x3_wmma<<<grid, 256, 0, stream>>>(f, x, Wc, bc, out);
}
